// MoE_40870908789399
// MI455X (gfx1250) — matrix operand order repaired and checked by reading, not compiled, not run on hardware
//
#include <hip/hip_runtime.h>

#define N_TOK 4096
#define D_DIM 768
#define H_DIM 3072
#define N_EXP 8
#define NSLOT (N_TOK * 2)

#define MT  32             // tokens per block (2 sub-tiles of 16)
#define HC  128            // H chunk per fused iteration (8 waves x 16 cols)
#define LDX (D_DIM + 8)    // bf16 LDS stride for x tile (bank spread: 388dw%64=4)
#define LDH (HC + 8)       // bf16 LDS stride for activation tile (68dw%64=4)

typedef __attribute__((ext_vector_type(16))) __bf16 v16bf;
typedef __attribute__((ext_vector_type(8)))  float  v8f;

// ---------------------------------------------------------------- init
__global__ __launch_bounds__(256) void moe_init_kernel(float* __restrict__ out,
                                                       unsigned int* cnt,
                                                       float* probsum) {
  size_t gid = (size_t)blockIdx.x * 256 + threadIdx.x;
  size_t b = gid * 4;
#pragma unroll
  for (int i = 0; i < 4; ++i) {
    size_t j = b + i;
    if (j < (size_t)N_TOK * D_DIM) out[j] = 0.f;
  }
  if (gid < N_EXP) { cnt[gid] = 0u; probsum[gid] = 0.f; }
}

// ---------------------------------------------------------------- gating
__global__ __launch_bounds__(256) void moe_gate_kernel(
    const float* __restrict__ x, const float* __restrict__ gw,
    unsigned int* cnt, float* probsum, int* tok_e, float* tok_w) {
  int n = blockIdx.x;
  int tid = threadIdx.x;
  int wv = tid >> 5, lane = tid & 31;
  __shared__ float logits[N_EXP];

  const float* xr = x + (size_t)n * D_DIM;
  const float* gr = gw + (size_t)wv * D_DIM;
  float s = 0.f;
  for (int d = lane; d < D_DIM; d += 32) s += xr[d] * gr[d];
#pragma unroll
  for (int m = 16; m >= 1; m >>= 1) s += __shfl_xor(s, m, 32);
  if (lane == 0) logits[wv] = s;
  __syncthreads();

  if (tid == 0) {
    float p[N_EXP];
    float mx = logits[0];
#pragma unroll
    for (int e = 1; e < N_EXP; ++e) mx = fmaxf(mx, logits[e]);
    float sum = 0.f;
#pragma unroll
    for (int e = 0; e < N_EXP; ++e) { p[e] = expf(logits[e] - mx); sum += p[e]; }
    float inv = 1.f / sum;
#pragma unroll
    for (int e = 0; e < N_EXP; ++e) { p[e] *= inv; atomicAdd(&probsum[e], p[e]); }
    // stable top-2 (first occurrence wins ties, like jax.lax.top_k)
    int i0 = 0;
#pragma unroll
    for (int e = 1; e < N_EXP; ++e) if (p[e] > p[i0]) i0 = e;
    int i1 = (i0 == 0) ? 1 : 0;
#pragma unroll
    for (int e = 0; e < N_EXP; ++e) if (e != i0 && p[e] > p[i1]) i1 = e;
    float w0 = p[i0], w1 = p[i1];
    float inv2 = 1.f / (w0 + w1);
    tok_e[2 * n] = i0;        tok_e[2 * n + 1] = i1;
    tok_w[2 * n] = w0 * inv2; tok_w[2 * n + 1] = w1 * inv2;
    atomicAdd(&cnt[i0], 1u);
    atomicAdd(&cnt[i1], 1u);
  }
}

// ---------------------------------------------------------------- offsets + balance loss
__global__ void moe_offsets_kernel(const unsigned int* cnt, unsigned int* cursor,
                                   unsigned int* offs, const float* probsum,
                                   float* balance_out) {
  if (threadIdx.x == 0 && blockIdx.x == 0) {
    unsigned int o = 0;
    float bal = 0.f;
#pragma unroll
    for (int e = 0; e < N_EXP; ++e) {
      cursor[e] = o; offs[e] = o; o += cnt[e];
      bal += (probsum[e] / (float)N_TOK) * ((float)cnt[e] / (float)N_TOK);
    }
    balance_out[0] = bal * (float)N_EXP;
  }
}

// ---------------------------------------------------------------- scatter into per-expert lists
__global__ __launch_bounds__(256) void moe_scatter_kernel(
    const int* __restrict__ tok_e, const float* __restrict__ tok_w,
    unsigned int* cursor, int* perm_tok, float* perm_w) {
  int s = blockIdx.x * 256 + threadIdx.x;
  if (s < NSLOT) {
    int e = tok_e[s];
    unsigned int pos = atomicAdd(&cursor[e], 1u);
    perm_tok[pos] = s >> 1;
    perm_w[pos]   = tok_w[s];
  }
}

// ---------------------------------------------------------------- fused expert MLP (bf16 WMMA)
// Block = 32 gathered tokens of one expert, 8 waves. Each Wfc/Wproj B-fragment
// is loaded once and reused for both 16-row M sub-tiles (2x arithmetic
// intensity per weight byte vs a 16-token tile).
__global__ __launch_bounds__(256) void moe_mlp_kernel(
    const float* __restrict__ x, const float* __restrict__ w_fc,
    const float* __restrict__ w_proj, const unsigned int* __restrict__ cnt,
    const unsigned int* __restrict__ offs, const int* __restrict__ perm_tok,
    const float* __restrict__ perm_w, float* __restrict__ out) {
  const int e = blockIdx.y;
  const unsigned int ce = cnt[e];
  const unsigned int base = blockIdx.x * (unsigned int)MT;
  if (base >= ce) return;

  const int tid = threadIdx.x;
  const int wv = tid >> 5, lane = tid & 31;

  __shared__ __bf16 xs[MT * LDX];
  __shared__ __bf16 hs[MT * LDH];
  __shared__ int   toks[MT];
  __shared__ float wts[MT];

  if (tid < MT) {
    unsigned int idx = base + (unsigned int)tid;
    bool valid = idx < ce;
    unsigned int src = offs[e] + (valid ? idx : base);
    toks[tid] = perm_tok[src];
    wts[tid]  = valid ? perm_w[src] : 0.f;   // padded rows contribute 0
  }
  __syncthreads();

  for (int i = tid; i < MT * D_DIM; i += 256) {
    int r = i / D_DIM, c = i - r * D_DIM;
    xs[r * LDX + c] = (__bf16)x[(size_t)toks[r] * D_DIM + c];
  }
  __syncthreads();

  const float* wfc_e = w_fc   + (size_t)e * H_DIM * D_DIM;  // [H,D]
  const float* wpr_e = w_proj + (size_t)e * D_DIM * H_DIM;  // [D,H]

  v8f acc0[6], acc1[6];
#pragma unroll
  for (int t = 0; t < 6; ++t) {
    acc0[t] = (v8f){0.f,0.f,0.f,0.f,0.f,0.f,0.f,0.f};
    acc1[t] = (v8f){0.f,0.f,0.f,0.f,0.f,0.f,0.f,0.f};
  }

  // ISA 7.12.2 fragment coordinates (wave32)
  const int arow = lane & 15;              // A: M row within sub-tile
  const int akh  = (lane >> 4) << 3;       // A: K half offset (0/8, +16 for hi VGPRs)
  const int bn   = lane & 15;              // B: N col
  const int bkh  = (lane >> 4) << 3;       // B: K half offset (0/8, +16 for hi VGPRs)
  const int moff = (lane >> 4) << 3;       // C/D: M = moff + v
  const int ncol = lane & 15;              // C/D: N col

  for (int hc = 0; hc < H_DIM; hc += HC) {
    // ---- stage A: h = relu(x @ Wfc^T)^2; wave owns 16 H-cols, both M sub-tiles
    const int hcol = hc + wv * 16;
    v8f ha0 = (v8f){0.f,0.f,0.f,0.f,0.f,0.f,0.f,0.f};
    v8f ha1 = (v8f){0.f,0.f,0.f,0.f,0.f,0.f,0.f,0.f};
    for (int kb = 0; kb < D_DIM; kb += 32) {
      v16bf a0, a1, b;
      const float* pb = wfc_e + (size_t)(hcol + bn) * D_DIM + kb + bkh;
#pragma unroll
      for (int i = 0; i < 8; ++i) { b[i] = (__bf16)pb[i]; b[8 + i] = (__bf16)pb[16 + i]; }
      const __bf16* pa0 = &xs[arow * LDX + kb + akh];
      const __bf16* pa1 = &xs[(16 + arow) * LDX + kb + akh];
#pragma unroll
      for (int i = 0; i < 8; ++i) { a0[i] = pa0[i]; a0[8 + i] = pa0[16 + i]; }
#pragma unroll
      for (int i = 0; i < 8; ++i) { a1[i] = pa1[i]; a1[8 + i] = pa1[16 + i]; }
      ha0 = __builtin_amdgcn_wmma_f32_16x16x32_bf16(false, a0, false, b,
                                                    (short)0, ha0, false, false);
      ha1 = __builtin_amdgcn_wmma_f32_16x16x32_bf16(false, a1, false, b,
                                                    (short)0, ha1, false, false);
    }
#pragma unroll
    for (int v = 0; v < 8; ++v) {
      float h0 = fmaxf(ha0[v], 0.f); h0 *= h0;
      float h1 = fmaxf(ha1[v], 0.f); h1 *= h1;
      hs[(moff + v) * LDH + wv * 16 + ncol]        = (__bf16)h0;
      hs[(16 + moff + v) * LDH + wv * 16 + ncol]   = (__bf16)h1;
    }
    __syncthreads();

    // ---- stage B: out_acc += h_chunk @ Wproj^T; wave owns 6 D-tiles (96 cols),
    //      B fragment shared by both M sub-tiles
#pragma unroll
    for (int t = 0; t < 6; ++t) {
      const int dcol = wv * 96 + t * 16;
#pragma unroll
      for (int kk = 0; kk < HC; kk += 32) {
        v16bf a0, a1, b;
        const float* pb = wpr_e + (size_t)(dcol + bn) * H_DIM + hc + kk + bkh;
#pragma unroll
        for (int i = 0; i < 8; ++i) { b[i] = (__bf16)pb[i]; b[8 + i] = (__bf16)pb[16 + i]; }
        const __bf16* pa0 = &hs[arow * LDH + kk + akh];
        const __bf16* pa1 = &hs[(16 + arow) * LDH + kk + akh];
#pragma unroll
        for (int i = 0; i < 8; ++i) { a0[i] = pa0[i]; a0[8 + i] = pa0[16 + i]; }
#pragma unroll
        for (int i = 0; i < 8; ++i) { a1[i] = pa1[i]; a1[8 + i] = pa1[16 + i]; }
        acc0[t] = __builtin_amdgcn_wmma_f32_16x16x32_bf16(false, a0, false, b,
                                                          (short)0, acc0[t], false, false);
        acc1[t] = __builtin_amdgcn_wmma_f32_16x16x32_bf16(false, a1, false, b,
                                                          (short)0, acc1[t], false, false);
      }
    }
    __syncthreads();
  }

  // routed-weighted scatter-add (each token is in exactly 2 expert lists)
#pragma unroll
  for (int t = 0; t < 6; ++t) {
    const int dcol = wv * 96 + t * 16 + ncol;
#pragma unroll
    for (int v = 0; v < 8; ++v) {
      int m0 = moff + v;
      int m1 = 16 + moff + v;
      atomicAdd(&out[(size_t)toks[m0] * D_DIM + dcol], wts[m0] * acc0[t][v]);
      atomicAdd(&out[(size_t)toks[m1] * D_DIM + dcol], wts[m1] * acc1[t][v]);
    }
  }
}

// ---------------------------------------------------------------- launch
extern "C" void kernel_launch(void* const* d_in, const int* in_sizes, int n_in,
                              void* d_out, int out_size, void* d_ws, size_t ws_size,
                              hipStream_t stream) {
  const float* x      = (const float*)d_in[0];
  const float* gate_w = (const float*)d_in[1];
  const float* w_fc   = (const float*)d_in[2];
  const float* w_proj = (const float*)d_in[3];
  float* out = (float*)d_out;

  char* ws = (char*)d_ws;                       // ~131 KB total
  unsigned int* cnt    = (unsigned int*)(ws);
  unsigned int* cursor = (unsigned int*)(ws + 32);
  unsigned int* offs   = (unsigned int*)(ws + 64);
  float* probsum       = (float*)(ws + 96);
  int*   tok_e         = (int*)  (ws + 128);
  float* tok_w         = (float*)(ws + 128 + 4 * NSLOT);
  int*   perm_tok      = (int*)  (ws + 128 + 8 * NSLOT);
  float* perm_w        = (float*)(ws + 128 + 12 * NSLOT);

  moe_init_kernel<<<(N_TOK * D_DIM / 4 + 255) / 256, 256, 0, stream>>>(out, cnt, probsum);
  moe_gate_kernel<<<N_TOK, 256, 0, stream>>>(x, gate_w, cnt, probsum, tok_e, tok_w);
  moe_offsets_kernel<<<1, 32, 0, stream>>>(cnt, cursor, offs, probsum,
                                           out + (size_t)N_TOK * D_DIM);
  moe_scatter_kernel<<<NSLOT / 256, 256, 0, stream>>>(tok_e, tok_w, cursor,
                                                      perm_tok, perm_w);
  dim3 grid((NSLOT + MT - 1) / MT, N_EXP);
  moe_mlp_kernel<<<grid, 256, 0, stream>>>(x, w_fc, w_proj, cnt, offs,
                                           perm_tok, perm_w, out);
}
